// CP_TRL_46591805227672
// MI455X (gfx1250) — matrix operand order repaired and checked by reading, not compiled, not run on hardware
//
#include <hip/hip_runtime.h>

typedef __attribute__((ext_vector_type(16))) _Float16 v16h;
typedef __attribute__((ext_vector_type(8)))  float    v8f;

#define NB   256      // batch
#define M1_  48
#define M2_  48
#define K_   48       // M3, contracted by WMMA (padded to 64)
#define C_   1000
#define R_   64
#define P_   (M1_*M2_)        // 2304 rows of x[b] viewed as (P_ x K_)
#define ROWTILES (P_/16)      // 144

// ---------------------------------------------------------------------------
// Kernel 1: per-batch fused contraction
//   tw[b][r] = weight[r] * sum_{i,j} ( sum_k x[b,i,j,k]*f2[k,r] ) * f0[i,r]*f1[j,r]
// GEMM part runs on v_wmma_f32_16x16x32_f16 (f16 in, f32 accumulate).
// ---------------------------------------------------------------------------
__global__ __launch_bounds__(256) void cp_contract_kernel(
    const float* __restrict__ x,       // [NB][M1][M2][K_]
    const float* __restrict__ f0,      // [M1][R_]
    const float* __restrict__ f1,      // [M2][R_]
    const float* __restrict__ f2,      // [K_][R_]
    const float* __restrict__ weight,  // [R_]
    float* __restrict__ tw)            // [NB][R_]  (workspace)
{
    __shared__ float f0s[M1_ * R_];
    __shared__ float f1s[M2_ * R_];
    __shared__ float ts[R_];

    const int b   = blockIdx.x;
    const int tid = threadIdx.x;
    const float* xb = x + (size_t)b * (P_ * K_);

    for (int idx = tid; idx < M1_ * R_; idx += 256) f0s[idx] = f0[idx];
    for (int idx = tid; idx < M2_ * R_; idx += 256) f1s[idx] = f1[idx];
    if (tid < R_) ts[tid] = 0.0f;
    __syncthreads();

    const int lane = tid & 31;
    const int wave = tid >> 5;
    const int hi   = lane >> 4;    // which half of the wave
    const int ln   = lane & 15;

    // ---- Preload B fragments: f2 (48x64 row-major), K padded 48->64 with 0.
    // 16-bit B 32x16 layout: half h of lane -> K = kp*32 + (h<8 ? h : h+8) + 8*hi, N = ln.
    v16h bfrag[4][2];
#pragma unroll
    for (int nt = 0; nt < 4; ++nt) {
#pragma unroll
        for (int kp = 0; kp < 2; ++kp) {
#pragma unroll
            for (int h = 0; h < 16; ++h) {
                const int k = kp * 32 + ((h < 8) ? h : h + 8) + 8 * hi;
                const int r = nt * 16 + ln;
                const float v = (k < K_) ? f2[k * R_ + r] : 0.0f;
                bfrag[nt][kp][h] = (_Float16)v;
            }
        }
    }

    float tacc[4] = {0.f, 0.f, 0.f, 0.f};

    for (int tile = wave; tile < ROWTILES; tile += 8) {
        const int rowbase = tile * 16;
        const float* rowp = xb + (rowbase + ln) * K_;   // lane's A row (M = ln)

        // Prefetch this wave's next tile row into cache (gfx1250 global_prefetch_b8)
        if (tile + 8 < ROWTILES)
            __builtin_prefetch(rowp + 8 * 16 * K_, 0, 0);

        // ---- Build A fragments (16-bit A 16x32 layout):
        // halves 0..7  -> K = kb .. kb+7,  halves 8..15 -> K = kb+16 .. kb+23,
        // kb = kp*32 + 8*hi.  kp=1 upper group is the zero padding (K>=48).
        v16h afrag[2];
#pragma unroll
        for (int kp = 0; kp < 2; ++kp) {
            const int kb = kp * 32 + 8 * hi;
            const float4 ga = *(const float4*)(rowp + kb);
            const float4 gb = *(const float4*)(rowp + kb + 4);
            afrag[kp][0] = (_Float16)ga.x;  afrag[kp][1] = (_Float16)ga.y;
            afrag[kp][2] = (_Float16)ga.z;  afrag[kp][3] = (_Float16)ga.w;
            afrag[kp][4] = (_Float16)gb.x;  afrag[kp][5] = (_Float16)gb.y;
            afrag[kp][6] = (_Float16)gb.z;  afrag[kp][7] = (_Float16)gb.w;
            if (kp == 0) {
                const float4 gc = *(const float4*)(rowp + kb + 16);
                const float4 gd = *(const float4*)(rowp + kb + 20);
                afrag[kp][8]  = (_Float16)gc.x;  afrag[kp][9]  = (_Float16)gc.y;
                afrag[kp][10] = (_Float16)gc.z;  afrag[kp][11] = (_Float16)gc.w;
                afrag[kp][12] = (_Float16)gd.x;  afrag[kp][13] = (_Float16)gd.y;
                afrag[kp][14] = (_Float16)gd.z;  afrag[kp][15] = (_Float16)gd.w;
            } else {
#pragma unroll
                for (int h = 8; h < 16; ++h) afrag[kp][h] = (_Float16)0.0f;
            }
        }

        // ---- 4 N-tiles x 2 K-passes of WMMA, then weighted reduction.
#pragma unroll
        for (int nt = 0; nt < 4; ++nt) {
            v8f cacc = {};
            cacc = __builtin_amdgcn_wmma_f32_16x16x32_f16(
                false, afrag[0], false, bfrag[nt][0], (short)0, cacc, false, false);
            cacc = __builtin_amdgcn_wmma_f32_16x16x32_f16(
                false, afrag[1], false, bfrag[nt][1], (short)0, cacc, false, false);

            const int r = nt * 16 + ln;
            float s = 0.0f;
#pragma unroll
            for (int d = 0; d < 8; ++d) {
                const int p = rowbase + d + 8 * hi;   // C layout: M = d + 8*hi
                const int i = p / M2_;
                const int j = p - i * M2_;
                s += cacc[d] * f0s[i * R_ + r] * f1s[j * R_ + r];
            }
            tacc[nt] += s;
        }
    }

#pragma unroll
    for (int nt = 0; nt < 4; ++nt)
        atomicAdd(&ts[nt * 16 + ln], tacc[nt]);   // ds_add_f32
    __syncthreads();

    if (tid < R_)
        tw[b * R_ + tid] = ts[tid] * weight[tid];
}

// ---------------------------------------------------------------------------
// Kernel 2: out[b][c] = sum_r tw[b][r] * f3[c][r] + bias   (256x64 @ 64x1000)
// One wave per 16x16 output tile; K=64 via two f16 WMMAs.
// ---------------------------------------------------------------------------
__global__ __launch_bounds__(256) void cp_out_kernel(
    const float* __restrict__ tw,    // [NB][R_]
    const float* __restrict__ f3,    // [C_][R_]
    const float* __restrict__ bias,  // [1]
    float* __restrict__ out)         // [NB][C_]
{
    const int tid  = threadIdx.x;
    const int lane = tid & 31;
    const int w    = blockIdx.x * 8 + (tid >> 5);  // 0..1007
    const int mt   = w & 15;        // 16 row tiles (256 rows)
    const int nt   = w >> 4;        // 63 col tiles (1000 cols, padded)
    const int hi   = lane >> 4;
    const int ln   = lane & 15;

    // A fragment: tw row, 16-bit A 16x32 layout
    const float* trow = tw + (mt * 16 + ln) * R_;
    v16h a[2];
#pragma unroll
    for (int kp = 0; kp < 2; ++kp) {
#pragma unroll
        for (int h = 0; h < 16; ++h) {
            const int k = kp * 32 + ((h < 8) ? h : h + 8) + 8 * hi;
            a[kp][h] = (_Float16)trow[k];
        }
    }

    // B fragment: B[k][n] = f3[c][k] (f3 rows are contiguous over k)
    const int  c  = nt * 16 + ln;
    const bool cv = (c < C_);
    const float* f3row = f3 + (cv ? c : 0) * R_;
    v16h bf[2];
#pragma unroll
    for (int kp = 0; kp < 2; ++kp) {
#pragma unroll
        for (int h = 0; h < 16; ++h) {
            const int k = kp * 32 + ((h < 8) ? h : h + 8) + 8 * hi;
            bf[kp][h] = (_Float16)(cv ? f3row[k] : 0.0f);
        }
    }

    v8f acc = {};
    acc = __builtin_amdgcn_wmma_f32_16x16x32_f16(
        false, a[0], false, bf[0], (short)0, acc, false, false);
    acc = __builtin_amdgcn_wmma_f32_16x16x32_f16(
        false, a[1], false, bf[1], (short)0, acc, false, false);

    const float bb = bias[0];
    if (cv) {
#pragma unroll
        for (int d = 0; d < 8; ++d) {
            const int row = mt * 16 + d + 8 * hi;
            out[row * C_ + c] = acc[d] + bb;
        }
    }
}

// ---------------------------------------------------------------------------
extern "C" void kernel_launch(void* const* d_in, const int* in_sizes, int n_in,
                              void* d_out, int out_size, void* d_ws, size_t ws_size,
                              hipStream_t stream) {
    const float* x      = (const float*)d_in[0];
    const float* weight = (const float*)d_in[1];
    const float* f0     = (const float*)d_in[2];
    const float* f1     = (const float*)d_in[3];
    const float* f2     = (const float*)d_in[4];
    const float* f3     = (const float*)d_in[5];
    const float* bias   = (const float*)d_in[6];
    float* out = (float*)d_out;
    float* tw  = (float*)d_ws;   // NB * R_ floats = 64 KB scratch

    cp_contract_kernel<<<NB, 256, 0, stream>>>(x, f0, f1, f2, weight, tw);
    cp_out_kernel<<<126, 256, 0, stream>>>(tw, f3, bias, out);
}
